// PrimaryCap_77154792506128
// MI455X (gfx1250) — matrix operand order repaired and checked by reading, not compiled, not run on hardware
//
#include <hip/hip_runtime.h>

// ---------------------------------------------------------------------------
// PrimaryCaps for MI455X (gfx1250, wave32, WMMA).
//   conv3x3 s2 SAME (im2col GEMM, f16 WMMA 16x16x32, f32 acc)
//   + bias + relu + per-group 16x16 CapsAct matmul (fused, 2nd WMMA)
//   + bias + [B,Ho,Wo,e,n] transpose via LDS, coalesced stores.
// Workspace layout (needs ~53.8 MB):
//   [0)                in16  : input cast to f16, 25,690,112 halfs
//   [51,380,224)       Bt    : conv weights transposed to [N=512][K=2304] f16
//   [53,739,520)       WcapA : CapsAct weights in WMMA-A per-lane order
// ---------------------------------------------------------------------------

typedef __attribute__((ext_vector_type(16))) _Float16 v16h;
typedef __attribute__((ext_vector_type(8)))  _Float16 v8h;
typedef __attribute__((ext_vector_type(8)))  float    v8f;

union V16 { v16h v; v8h h[2]; };

#define KTOT  2304
#define NCH   512
#define CHUNK 768          // K per LDS chunk = one ky row (3*256)
#define AROW  776          // padded LDS A row stride (halfs): 768+8 breaks bank stride
#define OROW  516          // padded LDS out row stride (floats): 512+4

#if defined(__HIP_DEVICE_COMPILE__) && defined(__gfx1250__) && \
    __has_builtin(__builtin_amdgcn_global_load_async_to_lds_b128)
#define HAVE_ASYNC_LDS 1
// Builtin prototype (from clang diagnostic): param0 = AS1 pointer to
// gcc-vector int4, param1 = AS3 pointer to int4, then imm offset, imm cpol.
typedef int v4i_g __attribute__((vector_size(16)));
typedef __attribute__((address_space(1))) v4i_g as1_v4i;
typedef __attribute__((address_space(3))) v4i_g as3_v4i;
#else
#define HAVE_ASYNC_LDS 0
#endif

__global__ __launch_bounds__(256) void k_cvt_input(const float* __restrict__ in,
                                                   _Float16* __restrict__ out) {
  size_t gid = (size_t)blockIdx.x * 256 + threadIdx.x;   // 3,211,264 threads * 8 elems
  const float4* p4 = (const float4*)in;
  float4 a = p4[gid * 2 + 0];
  float4 b = p4[gid * 2 + 1];
  v8h o;
  o[0] = (_Float16)a.x; o[1] = (_Float16)a.y; o[2] = (_Float16)a.z; o[3] = (_Float16)a.w;
  o[4] = (_Float16)b.x; o[5] = (_Float16)b.y; o[6] = (_Float16)b.z; o[7] = (_Float16)b.w;
  *(v8h*)(out + gid * 8) = o;
}

// Bt[n][k] = convW[k][n]  (HWIO flattens to exactly im2col K order)
__global__ __launch_bounds__(256) void k_cvt_conv_w(const float* __restrict__ w,
                                                    _Float16* __restrict__ bt) {
  int gid = blockIdx.x * 256 + threadIdx.x;              // < 512*2304
  int n = gid / KTOT;
  int k = gid - n * KTOT;
  bt[gid] = (_Float16)w[k * NCH + n];
}

// WcapA[g][lane][i] = CapsAct_W[din][g*16+e], e = lane&15, din = (lane>>4)*8 + i
// -> per-lane 16B = the low 8 halfs of the WMMA A fragment (K 16..31 zero-padded).
__global__ __launch_bounds__(256) void k_cvt_caps_w(const float* __restrict__ w,
                                                    _Float16* __restrict__ wa) {
  int gid = blockIdx.x * 256 + threadIdx.x;              // < 8192
  int g    = gid >> 8;
  int rem  = gid & 255;
  int lane = rem >> 3;
  int i    = rem & 7;
  int e    = lane & 15;
  int din  = ((lane >> 4) << 3) + i;
  wa[gid] = (_Float16)w[din * NCH + g * 16 + e];
}

__global__ __launch_bounds__(256) void k_primarycaps_main(
    const _Float16* __restrict__ in16,   // [32][56][56][256] f16
    const _Float16* __restrict__ Bt,     // [512][2304] f16
    const _Float16* __restrict__ WcapA,  // [32][32][8] f16
    const float*    __restrict__ bias1,  // [512]
    const float*    __restrict__ capsB,  // [512]
    float*          __restrict__ out)    // [25088][16][32]
{
  __shared__ __align__(16) _Float16 A_lds[16 * AROW];    // im2col patch chunk
  __shared__ __align__(16) float    out_lds[16 * OROW];  // [m][e*32+g] staging

  const int tid   = threadIdx.x;
  const int lane  = tid & 31;
  const int wv    = tid >> 5;       // wave 0..7
  const int half  = lane >> 4;      // 0/1
  const int li    = lane & 15;
  const int mtile = blockIdx.x;     // 1568 tiles of 16 spatial positions
  const int g0    = wv * 4;         // this wave's first capsule group

  v8f acc[4];
#pragma unroll
  for (int gi = 0; gi < 4; ++gi)
#pragma unroll
    for (int j = 0; j < 8; ++j) acc[gi][j] = 0.f;

  // ---- GEMM1 (transposed): D1[din][m] = sum_K Bt[g*16+din][K] * patch[m][K]
  for (int c = 0; c < 3; ++c) {     // chunk c == ky
    __syncthreads();                // previous chunk fully consumed

    // stage 16 rows x 3 (kx) x 256 cin f16 into LDS; 6 segments per wave,
    // each segment = 512 B = 32 lanes x 16 B (async DMA to LDS when available)
#pragma unroll
    for (int s6 = 0; s6 < 6; ++s6) {
      int s   = wv * 6 + s6;        // 0..47
      int r   = s / 3;              // tile row
      int pp  = s - r * 3;          // kx
      int m   = mtile * 16 + r;
      int b   = m / 784;  int rem = m - b * 784;
      int ho  = rem / 28; int wo  = rem - ho * 28;
      int ih  = 2 * ho + c  - 1;
      int iw  = 2 * wo + pp - 1;
      _Float16* lp = &A_lds[r * AROW + pp * 256 + lane * 8];
      if ((unsigned)ih < 56u && (unsigned)iw < 56u) {
        const _Float16* gp = in16 + ((size_t)((b * 56 + ih) * 56 + iw) * 256 + lane * 8);
#if HAVE_ASYNC_LDS
        __builtin_amdgcn_global_load_async_to_lds_b128(
            (as1_v4i*)(void*)gp, (as3_v4i*)(void*)lp, 0, 0);
#else
        *(v8h*)lp = *(const v8h*)gp;
#endif
      } else {
        *(int4*)lp = make_int4(0, 0, 0, 0);   // SAME padding -> zeros
      }
    }
#if HAVE_ASYNC_LDS
#if __has_builtin(__builtin_amdgcn_s_wait_asynccnt)
    __builtin_amdgcn_s_wait_asynccnt(0);
#else
    asm volatile("s_wait_asynccnt 0x0" ::: "memory");
#endif
#endif
    __syncthreads();

    for (int t = 0; t < 24; ++t) {
      // B fragment (patchT, K x M): lane L: N=m=L&15, halfs0..7: K=k0+(L>>4)*8.., halfs8..15: K=k0+16+(L>>4)*8..
      const _Float16* bp = &A_lds[li * AROW + t * 32 + half * 8];
      V16 bf;
      bf.h[0] = *(const v8h*)bp;
      bf.h[1] = *(const v8h*)(bp + 16);
      const int kglob = c * CHUNK + t * 32;
#pragma unroll
      for (int gi = 0; gi < 4; ++gi) {
        // A fragment (weights, din x K): lane L: row din=L&15,
        //   halfs0..7: K=k0+(L>>4)*8.., halfs8..15: K=k0+16+(L>>4)*8..
        const _Float16* ap = Bt + (size_t)((g0 + gi) * 16 + li) * KTOT + kglob + half * 8;
        V16 af;
        af.h[0] = *(const v8h*)ap;
        af.h[1] = *(const v8h*)(ap + 16);
        acc[gi] = __builtin_amdgcn_wmma_f32_16x16x32_f16(
            false, af.v, false, bf.v, (short)0, acc[gi], false, false);
      }
    }
  }

  // ---- fused epilogue: bias + relu, then CapsAct WMMA (K=16 padded to 32),
  //      D2[e][m] = Wcap[e][din] * relu(D1)[din][m], then caps bias.
#pragma unroll
  for (int gi = 0; gi < 4; ++gi) {
    const int g = g0 + gi;
    float own[8];
#pragma unroll
    for (int j = 0; j < 8; ++j) {
      // D1 lane mapping: row din = j + 8*half, col m = li
      float x = acc[gi][j] + bias1[g * 16 + half * 8 + j];
      own[j] = x > 0.f ? x : 0.f;
    }
    // Build B2: lane L holds din (L>>4)*8 + 0..7 at m=L&15 in halfs 0..7.
    v16h b2;
#pragma unroll
    for (int j = 0; j < 8; ++j) {
      float oth = __shfl_xor(own[j], 16, 32);
      b2[j]     = (_Float16)own[j];
      b2[j + 8] = (_Float16)0.f;  // K 16..31 -> 0
    }
    v8h w8 = *(const v8h*)(WcapA + (size_t)(g * 32 + lane) * 8);
    v16h a2;
#pragma unroll
    for (int i = 0; i < 8; ++i) { a2[i] = w8[i]; a2[i + 8] = (_Float16)0.f; }
    v8f c2;
#pragma unroll
    for (int j = 0; j < 8; ++j) c2[j] = 0.f;
    c2 = __builtin_amdgcn_wmma_f32_16x16x32_f16(
        false, a2, false, b2, (short)0, c2, false, false);
#pragma unroll
    for (int j = 0; j < 8; ++j) {
      // D2 lane mapping: row e = j + 8*half, col m = li
      float e = c2[j] + capsB[g * 16 + half * 8 + j];
      out_lds[li * OROW + (half * 8 + j) * 32 + g] = e;
    }
  }
  __syncthreads();

  // ---- coalesced store: 16 rows x 512 floats, float4 per lane
#pragma unroll
  for (int u = 0; u < 8; ++u) {
    int fidx = u * 256 + tid;           // 0..2047 float4s in tile
    int r2   = fidx >> 7;               // row m in tile
    int cx   = fidx & 127;              // float4 column
    float4 val = *(const float4*)(&out_lds[r2 * OROW + cx * 4]);
    ((float4*)out)[(size_t)mtile * 2048 + r2 * 128 + cx] = val;
  }
}

extern "C" void kernel_launch(void* const* d_in, const int* in_sizes, int n_in,
                              void* d_out, int out_size, void* d_ws, size_t ws_size,
                              hipStream_t stream) {
  (void)in_sizes; (void)n_in; (void)out_size; (void)ws_size;
  const float* inputs = (const float*)d_in[0];
  const float* convW  = (const float*)d_in[1];
  const float* bias1  = (const float*)d_in[2];
  const float* capsW  = (const float*)d_in[3];
  const float* capsB  = (const float*)d_in[4];

  char* ws = (char*)d_ws;
  _Float16* in16  = (_Float16*)ws;                              // 51,380,224 B
  _Float16* Bt    = (_Float16*)(ws + 51380224);                 //  2,359,296 B
  _Float16* WcapA = (_Float16*)(ws + 51380224 + 2359296);       //     16,384 B

  k_cvt_input <<<12544, 256, 0, stream>>>(inputs, in16);
  k_cvt_conv_w<<< 4608, 256, 0, stream>>>(convW, Bt);
  k_cvt_caps_w<<<   32, 256, 0, stream>>>(capsW, WcapA);
  k_primarycaps_main<<<1568, 256, 0, stream>>>(in16, Bt, WcapA, bias1, capsB,
                                               (float*)d_out);
}
